// WindowAttention2_90683939487946
// MI455X (gfx1250) — matrix operand order repaired and checked by reading, not compiled, not run on hardware
//
#include <hip/hip_runtime.h>

typedef __attribute__((ext_vector_type(16))) __bf16 v16bf;
typedef __attribute__((ext_vector_type(8)))  float  v8f;

#define NGT   49
#define NTOK  50
#define CDIM  128
#define HEADS 4
#define DHEAD 32
#define NIMG  64
#define NTPI  3136
#define SCALE 0.17677669529663687f   // 32^-0.5

// native bf16 convert (lowers to v_cvt_*bf16_f32, RTNE in hardware)
__device__ __forceinline__ unsigned short f2bf(float f) {
    union { __bf16 h; unsigned short u; } c;
    c.h = (__bf16)f;
    return c.u;
}
__device__ __forceinline__ float bf2f(unsigned short u) {
    return __uint_as_float(((unsigned int)u) << 16);
}

union ABf { v16bf v; unsigned short u[16]; unsigned int d[8]; uint4 q[2]; };
union Cf  { v8f   v; float f[8]; };

// A fragment: 16x32 bf16, A row-major in LDS (ushort), stride ld elements.
// lanes 0-15 = M rows, K 0-15; lanes 16-31: K 16-31 (contiguous).
__device__ __forceinline__ v16bf loadA(const unsigned short* s, int ld,
                                       int mRow, int kCol, int lane) {
    ABf f;
    const unsigned short* p = s + (mRow + (lane & 15)) * ld + kCol;
    int kh = (lane >> 4) << 4;
#pragma unroll
    for (int v = 0; v < 8; ++v) {
        int kb = v * 2 + kh;
        f.u[2 * v]     = p[kb];
        f.u[2 * v + 1] = p[kb + 1];
    }
    return f.v;
}

// B fragment: 32x16 bf16, B row-major (KxN) in LDS (ushort), stride ld.
__device__ __forceinline__ v16bf loadB(const unsigned short* s, int ld,
                                       int kRow, int nCol, int lane) {
    ABf f;
    const unsigned short* p = s + (kRow + ((lane >> 4) << 4)) * ld + nCol + (lane & 15);
#pragma unroll
    for (int j = 0; j < 8; ++j) {
        f.u[2 * j]     = p[(2 * j) * ld];
        f.u[2 * j + 1] = p[(2 * j + 1) * ld];
    }
    return f.v;
}

// ---------------------------------------------------------------------------
// Pre-pack [Wq | Wkv] into WMMA B-fragment order, bf16.
// 96 tiles: tile t -> nt = t/4 (0..23), kk = t%4. Within a tile:
//   element e (0..511): lane = e>>4, idx = e&15
//   k = kk*32 + (lane>>4)*16 + idx ; n = colbase + (lane&15)
// ---------------------------------------------------------------------------
__global__ __launch_bounds__(256)
void pack_w_kernel(const float* __restrict__ Wq, const float* __restrict__ Wkv,
                   unsigned short* __restrict__ wpack)
{
    int t = blockIdx.x;            // 0..95
    int nt = t >> 2, kk = t & 3;
    const float* W; int ld, colbase;
    if (nt < 8)       { W = Wq;  ld = CDIM; colbase = nt * 16; }
    else if (nt < 16) { W = Wkv; ld = 256;  colbase = (nt - 8) * 16; }
    else              { W = Wkv; ld = 256;  colbase = 128 + (nt - 16) * 16; }
    for (int e = threadIdx.x; e < 512; e += 256) {
        int lane = e >> 4, idx = e & 15;
        int k = kk * 32 + ((lane >> 4) << 4) + idx;
        int n = colbase + (lane & 15);
        wpack[(size_t)t * 512 + e] = f2bf(W[(size_t)k * ld + n]);
    }
}

// ---------------------------------------------------------------------------
// Pre-expand relative-position bias into padded [H][64][64] bf16 table.
// ---------------------------------------------------------------------------
__global__ __launch_bounds__(256)
void expand_bias_kernel(const float* __restrict__ btab, const int* __restrict__ ridx,
                        unsigned short* __restrict__ bexp)
{
    int i = blockIdx.x * 256 + threadIdx.x;      // 0 .. 4*64*64-1
    if (i >= HEADS * 64 * 64) return;
    int h = i >> 12, row = (i >> 6) & 63, col = i & 63;
    float v = 0.f;
    if (row < NGT && col < NTOK)
        v = btab[ridx[row * NTOK + col] * HEADS + h];
    bexp[i] = f2bf(v);
}

// ---------------------------------------------------------------------------
// Fused window attention: one block per window (4096 blocks x 256 threads)
// ---------------------------------------------------------------------------
__global__ __launch_bounds__(256)
void win_attn_kernel(const float* __restrict__ x, const float* __restrict__ xavg,
                     const float* __restrict__ mask,
                     const unsigned short* __restrict__ wpack,
                     const unsigned short* __restrict__ bexp,
                     const float* __restrict__ bq,
                     const float* __restrict__ bkv,
                     float* __restrict__ out)
{
    extern __shared__ char smem[];
    unsigned short* sX  = (unsigned short*)(smem);           // 64 x 128 bf16 (x_c padded)
    unsigned short* sQ  = (unsigned short*)(smem + 16384);   // 64 x 128 bf16 (scaled Q)
    unsigned short* sKT = (unsigned short*)(smem + 32768);   // 128 x 64 bf16 (K^T)
    unsigned short* sV  = (unsigned short*)(smem + 49152);   // 64 x 128 bf16
    unsigned short* sSP = (unsigned short*)(smem + 65536);   // 4 x 64 x 64 bf16 (S then P)

    const int blk  = blockIdx.x;
    const int img  = blk >> 6;
    const int w    = blk & 63;
    const int tid  = threadIdx.x;
    const int wave = tid >> 5;
    const int lane = tid & 31;

    // ---- stage 1: stage x_c into LDS as bf16 (vectorized), pad to 64 rows ----
    for (int i = tid * 4; i < 64 * CDIM; i += 1024) {
        int row = i >> 7, c = i & 127;
        float4 v = make_float4(0.f, 0.f, 0.f, 0.f);
        if (row < NGT)
            v = *(const float4*)(x + ((size_t)blk * NGT + row) * CDIM + c);
        else if (row == NGT)
            v = *(const float4*)(xavg + img * CDIM + c);
        ushort4 h;
        h.x = f2bf(v.x); h.y = f2bf(v.y); h.z = f2bf(v.z); h.w = f2bf(v.w);
        *(ushort4*)(sX + i) = h;
    }
    __syncthreads();

    // ---- stage 2: QKV = Xc @ [Wq | Wkv] via WMMA (pre-packed B fragments) ----
    for (int nt = wave; nt < 24; nt += 8) {
        Cf acc[4];
#pragma unroll
        for (int m = 0; m < 4; ++m)
#pragma unroll
            for (int j = 0; j < 8; ++j) acc[m].f[j] = 0.f;

        int kind, colbase;
        if (nt < 8)       { kind = 0; colbase = nt * 16; }
        else if (nt < 16) { kind = 1; colbase = (nt - 8) * 16; }
        else              { kind = 2; colbase = 128 + (nt - 16) * 16; }

        for (int kk = 0; kk < 4; ++kk) {
            ABf b;
            const uint4* wp =
                (const uint4*)(wpack + (((size_t)nt * 4 + kk) * 512) + lane * 16);
            b.q[0] = wp[0];
            b.q[1] = wp[1];
#pragma unroll
            for (int mt = 0; mt < 4; ++mt) {
                v16bf a = loadA(sX, CDIM, mt * 16, kk * 32, lane);
                acc[mt].v = __builtin_amdgcn_wmma_f32_16x16x32_bf16(
                    false, a, false, b.v, (short)0, acc[mt].v, false, false);
            }
        }
        int colg = colbase + (lane & 15);
        int coll = (kind == 2) ? colg - 128 : colg;
        float bias = (kind == 0) ? bq[colg] : bkv[colg];
#pragma unroll
        for (int mt = 0; mt < 4; ++mt) {
#pragma unroll
            for (int r = 0; r < 8; ++r) {
                int row = mt * 16 + r + ((lane >> 4) << 3);
                float val = acc[mt].f[r] + bias;
                if (kind == 0)      sQ[row * CDIM + coll] = f2bf(val * SCALE);
                else if (kind == 1) sKT[coll * 64 + row]  = f2bf(val);
                else                sV[row * CDIM + coll] = f2bf(val);
            }
        }
    }
    __syncthreads();

    // ---- stage 3: S = Q @ K^T per head, + pre-expanded bias + mask ----
    for (int job = wave; job < 64; job += 8) {
        int h = job >> 4, rem = job & 15, mt = rem >> 2, nt = rem & 3;
        Cf acc;
#pragma unroll
        for (int j = 0; j < 8; ++j) acc.f[j] = 0.f;
        v16bf a = loadA(sQ, CDIM, mt * 16, h * DHEAD, lane);
        v16bf b = loadB(sKT + h * DHEAD * 64, 64, 0, nt * 16, lane);
        acc.v = __builtin_amdgcn_wmma_f32_16x16x32_bf16(
            false, a, false, b, (short)0, acc.v, false, false);
        int col = nt * 16 + (lane & 15);
#pragma unroll
        for (int r = 0; r < 8; ++r) {
            int row = mt * 16 + r + ((lane >> 4) << 3);
            float s = acc.f[r] + bf2f(bexp[(h * 64 + row) * 64 + col]);
            if (row < NGT && col < NTOK)
                s += mask[(w * NGT + row) * NTOK + col];
            sSP[(h * 64 + row) * 64 + col] = f2bf(s);
        }
    }
    __syncthreads();

    // ---- stage 4: softmax over valid 50 cols, P (bf16) in place, pads zero --
    {
        int h = tid >> 6, m = tid & 63;
        unsigned short* rowp = sSP + (h * 64 + m) * 64;
        if (m < NGT) {
            float mx = -1e30f;
            for (int j = 0; j < NTOK; ++j) mx = fmaxf(mx, bf2f(rowp[j]));
            float den = 0.f;
            for (int j = 0; j < NTOK; ++j) {
                float e = __expf(bf2f(rowp[j]) - mx);
                den += e;
                rowp[j] = f2bf(e);
            }
            float inv = 1.f / den;
            for (int j = 0; j < NTOK; ++j) rowp[j] = f2bf(bf2f(rowp[j]) * inv);
            for (int j = NTOK; j < 64; ++j) rowp[j] = 0;
        } else {
            for (int j = 0; j < 64; ++j) rowp[j] = 0;
        }
    }
    __syncthreads();

    // ---- stage 5: O = P @ V per head, write to global ----
    for (int job = wave; job < 32; job += 8) {
        int h = job >> 3, mt = (job >> 1) & 3, nt = job & 1;
        Cf acc;
#pragma unroll
        for (int j = 0; j < 8; ++j) acc.f[j] = 0.f;
#pragma unroll
        for (int kk = 0; kk < 2; ++kk) {
            v16bf a = loadA(sSP + h * 64 * 64, 64, mt * 16, kk * 32, lane);
            v16bf b = loadB(sV, CDIM, kk * 32, h * DHEAD + nt * 16, lane);
            acc.v = __builtin_amdgcn_wmma_f32_16x16x32_bf16(
                false, a, false, b, (short)0, acc.v, false, false);
        }
        int col = h * DHEAD + nt * 16 + (lane & 15);
#pragma unroll
        for (int r = 0; r < 8; ++r) {
            int row = mt * 16 + r + ((lane >> 4) << 3);
            if (row < NGT)
                out[((size_t)blk * NGT + row) * CDIM + col] = acc.f[r];
        }
    }
}

// ---------------------------------------------------------------------------
// Global-token path (algebraically collapsed; tiny FLOP count)
// ---------------------------------------------------------------------------
__global__ __launch_bounds__(128)
void global_q_kernel(const float* __restrict__ xavg, const float* __restrict__ Wq,
                     const float* __restrict__ bq, const float* __restrict__ Wkv,
                     const float* __restrict__ bkv,
                     float* __restrict__ qW, float* __restrict__ sconst)
{
    __shared__ float qa[CDIM];
    int b = blockIdx.x, t = threadIdx.x;
    float s = bq[t];
    for (int c = 0; c < CDIM; ++c) s += xavg[b * CDIM + c] * Wq[c * CDIM + t];
    qa[t] = s;
    __syncthreads();
    for (int idx = t; idx < CDIM * HEADS; idx += 128) {
        int c = idx >> 2, h = idx & 3;
        float acc = 0.f;
        for (int kd = 0; kd < DHEAD; ++kd)
            acc += qa[h * DHEAD + kd] * Wkv[c * 256 + h * DHEAD + kd];
        qW[(b * CDIM + c) * HEADS + h] = acc;
    }
    if (t < HEADS) {
        float acc = 0.f;
        for (int kd = 0; kd < DHEAD; ++kd) acc += qa[t * DHEAD + kd] * bkv[t * DHEAD + kd];
        sconst[b * HEADS + t] = acc;
    }
}

__global__ __launch_bounds__(256)
void global_score_kernel(const float* __restrict__ xtotal, const float* __restrict__ qW,
                         const float* __restrict__ sconst, float* __restrict__ scores)
{
    int gid = blockIdx.x * 256 + threadIdx.x;
    if (gid >= NIMG * NTPI) return;
    int b = gid / NTPI, m = gid - b * NTPI;
    const float* xr = xtotal + (size_t)gid * CDIM;
    const float* qw = qW + b * CDIM * HEADS;
    float s0 = 0.f, s1 = 0.f, s2 = 0.f, s3 = 0.f;
    for (int c = 0; c < CDIM; ++c) {
        float xv = xr[c];
        s0 += xv * qw[c * 4 + 0];
        s1 += xv * qw[c * 4 + 1];
        s2 += xv * qw[c * 4 + 2];
        s3 += xv * qw[c * 4 + 3];
    }
    size_t base = (size_t)b * HEADS * NTPI + m;
    scores[base + 0 * NTPI] = s0 + sconst[b * 4 + 0];
    scores[base + 1 * NTPI] = s1 + sconst[b * 4 + 1];
    scores[base + 2 * NTPI] = s2 + sconst[b * 4 + 2];
    scores[base + 3 * NTPI] = s3 + sconst[b * 4 + 3];
}

__global__ __launch_bounds__(128)
void global_out_kernel(const float* __restrict__ xtotal, const float* __restrict__ Wkv,
                       const float* __restrict__ bkv, const float* __restrict__ scores,
                       float* __restrict__ outg)
{
    __shared__ float red[128];
    __shared__ float pe[128];
    int b = blockIdx.x >> 2, h = blockIdx.x & 3, t = threadIdx.x;
    const float* sc = scores + (size_t)(b * HEADS + h) * NTPI;

    float mx = -1e30f;
    for (int m = t; m < NTPI; m += 128) mx = fmaxf(mx, sc[m]);
    red[t] = mx; __syncthreads();
    for (int s = 64; s > 0; s >>= 1) {
        if (t < s) red[t] = fmaxf(red[t], red[t + s]);
        __syncthreads();
    }
    mx = red[0]; __syncthreads();

    float den = 0.f;
    for (int m = t; m < NTPI; m += 128) den += __expf(sc[m] - mx);
    red[t] = den; __syncthreads();
    for (int s = 64; s > 0; s >>= 1) {
        if (t < s) red[t] += red[t + s];
        __syncthreads();
    }
    den = red[0]; __syncthreads();
    float inv = 1.f / den;

    // y[c] = sum_m p_m * x[m][c]; thread owns channel t
    float y = 0.f;
    for (int m0 = 0; m0 < NTPI; m0 += 128) {
        int cnt = min(128, NTPI - m0);
        if (t < cnt) pe[t] = __expf(sc[m0 + t] - mx) * inv;
        __syncthreads();
        for (int mm = 0; mm < cnt; ++mm)
            y += pe[mm] * xtotal[((size_t)b * NTPI + m0 + mm) * CDIM + t];
        __syncthreads();
    }
    red[t] = y;
    __syncthreads();

    if (t < DHEAD) {
        float o = bkv[CDIM + h * DHEAD + t];
        for (int c = 0; c < CDIM; ++c)
            o += red[c] * Wkv[c * 256 + CDIM + h * DHEAD + t];
        outg[b * CDIM + h * DHEAD + t] = o;
    }
}

extern "C" void kernel_launch(void* const* d_in, const int* in_sizes, int n_in,
                              void* d_out, int out_size, void* d_ws, size_t ws_size,
                              hipStream_t stream) {
    (void)in_sizes; (void)n_in; (void)out_size; (void)ws_size;
    const float* x      = (const float*)d_in[0];
    const float* xtotal = (const float*)d_in[1];
    const float* xavg   = (const float*)d_in[2];
    const float* mask   = (const float*)d_in[3];
    const float* Wq     = (const float*)d_in[4];
    const float* bq     = (const float*)d_in[5];
    const float* Wkv    = (const float*)d_in[6];
    const float* bkv    = (const float*)d_in[7];
    const float* btab   = (const float*)d_in[8];
    const int*   ridx   = (const int*)d_in[9];

    float* out  = (float*)d_out;
    float* outg = out + (size_t)4096 * NGT * CDIM;   // xavg_out after x_out

    float* ws = (float*)d_ws;
    unsigned short* wpack = (unsigned short*)ws;        // 96*512 bf16   = 24576 floats
    unsigned short* bexp  = (unsigned short*)(ws + 24576); // 4*64*64 bf16 = 8192 floats
    float* qW     = ws + 24576 + 8192;                  // 64*128*4 = 32768
    float* sconst = ws + 24576 + 8192 + 32768;          // 256
    float* scores = ws + 24576 + 8192 + 33024;          // 64*4*3136

    pack_w_kernel<<<96, 256, 0, stream>>>(Wq, Wkv, wpack);
    expand_bias_kernel<<<(HEADS * 64 * 64 + 255) / 256, 256, 0, stream>>>(btab, ridx, bexp);
    win_attn_kernel<<<4096, 256, 98304, stream>>>(x, xavg, mask, wpack, bexp,
                                                  bq, bkv, out);
    global_q_kernel<<<NIMG, 128, 0, stream>>>(xavg, Wq, bq, Wkv, bkv, qW, sconst);
    global_score_kernel<<<(NIMG * NTPI + 255) / 256, 256, 0, stream>>>(xtotal, qW,
                                                                       sconst, scores);
    global_out_kernel<<<NIMG * HEADS, 128, 0, stream>>>(xtotal, Wkv, bkv, scores, outg);
}
